// Non_Local_Block_27006754357719
// MI455X (gfx1250) — matrix operand order repaired and checked by reading, not compiled, not run on hardware
//
#include <hip/hip_runtime.h>

#define BATCH 4
#define CIN   256
#define CI    128
#define CO    256
#define NPOS  4096

typedef __attribute__((ext_vector_type(16))) __bf16 v16bf;
typedef __attribute__((ext_vector_type(8)))  float  v8f;

#define SCHED_FENCE() __builtin_amdgcn_sched_barrier(0)

union FragAB {
  v16bf v;
  uint4 q[2];
  unsigned int u[8];
};

__device__ __forceinline__ unsigned short f2bf(float f) {
  union { __bf16 h; unsigned short s; } t;
  t.h = (__bf16)f;                       // hardware cvt, RNE
  return t.s;
}
__device__ __forceinline__ unsigned int pack_bf2(float a, float b) {
  union { __bf16 h[2]; unsigned int u; } t;
  t.h[0] = (__bf16)a;
  t.h[1] = (__bf16)b;
  return t.u;
}

// -----------------------------------------------------------------------------
// Kernel 1: theta/phi/g projections via bf16 WMMA.
//   thetaT, phiT : bf16 [B][N][CI]   (c-contiguous -> A/B fragment pair loads)
//   gmat         : bf16 [B][CI][N]   (n-contiguous -> B fragment pair loads)
// Workgroup: batch b x 128-position tile. 8 waves, wave = 16-channel block mb.
// kb-outer / nb-inner: 8 independent accumulator chains, one live W fragment.
// -----------------------------------------------------------------------------
__launch_bounds__(256)
__global__ void proj_kernel(const float* __restrict__ x,
                            const float* __restrict__ wt, const float* __restrict__ bt,
                            const float* __restrict__ wp, const float* __restrict__ bp,
                            const float* __restrict__ wg, const float* __restrict__ bg,
                            unsigned short* __restrict__ thetaT,
                            unsigned short* __restrict__ phiT,
                            unsigned short* __restrict__ gmat) {
  __shared__ __align__(16) unsigned short xT[128][CIN];   // 64 KB: [pos][cin] bf16

  const int b    = blockIdx.y;
  const int n0   = blockIdx.x * 128;
  const int tid  = threadIdx.x;
  const int wave = tid >> 5;
  const int lane = tid & 31;
  const int h    = lane >> 4;
  const int l15  = lane & 15;

  // Stage x tile: each wave streams whole cin rows (coalesced b128), transposed into LDS.
  for (int r = wave; r < CIN; r += 8) {
    const float4 f4 = *(const float4*)(x + (size_t)(b * CIN + r) * NPOS + n0 + lane * 4);
    xT[lane * 4 + 0][r] = f2bf(f4.x);
    xT[lane * 4 + 1][r] = f2bf(f4.y);
    xT[lane * 4 + 2][r] = f2bf(f4.z);
    xT[lane * 4 + 3][r] = f2bf(f4.w);
  }
  __syncthreads();

  const int mb = wave;                                   // 16-channel output block
  const char* xbase = (const char*)&xT[l15][0] + h * 16; // frag(nb,kb) = +nb*8192 +kb*64

  for (int p = 0; p < 3; ++p) {
    const float* wsel = (p == 0) ? wt : (p == 1) ? wp : wg;
    const float* bsel = (p == 0) ? bt : (p == 1) ? bp : bg;
    const float* wrow = wsel + (size_t)(mb * 16 + l15) * CIN;

    v8f acc[8];
    #pragma unroll
    for (int nb = 0; nb < 8; ++nb)
      #pragma unroll
      for (int r = 0; r < 8; ++r) acc[nb][r] = 0.f;

    FragAB aWc, aWn;
    {  // prologue gather: W A-fragment for kb=0 (f32 -> bf16 pack)
      const float* f0 = wrow + h * 8;
      const float* f1 = f0 + 16;
      #pragma unroll
      for (int j = 0; j < 4; ++j) {
        aWc.u[j]     = pack_bf2(f0[2 * j], f0[2 * j + 1]);
        aWc.u[4 + j] = pack_bf2(f1[2 * j], f1[2 * j + 1]);
      }
    }

    #pragma unroll
    for (int kb = 0; kb < 8; ++kb) {
      if (kb < 7) {            // prefetch next W fragment; hidden under 8 WMMAs
        const float* f0 = wrow + (kb + 1) * 32 + h * 8;
        const float* f1 = f0 + 16;
        #pragma unroll
        for (int j = 0; j < 4; ++j) {
          aWn.u[j]     = pack_bf2(f0[2 * j], f0[2 * j + 1]);
          aWn.u[4 + j] = pack_bf2(f1[2 * j], f1[2 * j + 1]);
        }
      }
      FragAB bc, bn;
      bc.q[0] = *(const uint4*)(xbase + kb * 64);
      bc.q[1] = *(const uint4*)(xbase + kb * 64 + 32);
      #pragma unroll
      for (int nb = 0; nb < 8; ++nb) {
        if (nb < 7) {
          const char* a2 = xbase + (nb + 1) * 8192 + kb * 64;
          bn.q[0] = *(const uint4*)(a2);
          bn.q[1] = *(const uint4*)(a2 + 32);
        }
        acc[nb] = __builtin_amdgcn_wmma_f32_16x16x32_bf16(false, aWc.v, false, bc.v,
                                                          (short)0, acc[nb], false, false);
        bc = bn;
        SCHED_FENCE();         // bound liveness: cur+next fragment only
      }
      aWc = aWn;
    }

    // ---- stores ----
    #pragma unroll
    for (int nb = 0; nb < 8; ++nb) {
      const int npos = n0 + nb * 16 + l15;
      if (p < 2) {
        unsigned short* dstT = (p == 0) ? thetaT : phiT;
        uint4 pk;
        pk.x = pack_bf2(acc[nb][0] + bsel[mb * 16 + 8 * h + 0], acc[nb][1] + bsel[mb * 16 + 8 * h + 1]);
        pk.y = pack_bf2(acc[nb][2] + bsel[mb * 16 + 8 * h + 2], acc[nb][3] + bsel[mb * 16 + 8 * h + 3]);
        pk.z = pack_bf2(acc[nb][4] + bsel[mb * 16 + 8 * h + 4], acc[nb][5] + bsel[mb * 16 + 8 * h + 5]);
        pk.w = pack_bf2(acc[nb][6] + bsel[mb * 16 + 8 * h + 6], acc[nb][7] + bsel[mb * 16 + 8 * h + 7]);
        *(uint4*)(dstT + (size_t)(b * NPOS + npos) * CI + mb * 16 + 8 * h) = pk;
      } else {
        #pragma unroll
        for (int r = 0; r < 8; ++r) {
          const int ch = mb * 16 + 8 * h + r;
          gmat[(size_t)(b * CI + ch) * NPOS + npos] = f2bf(acc[nb][r] + bsel[ch]);
        }
      }
    }
  }
}

// -----------------------------------------------------------------------------
// Kernel 2: fused flash attention. Workgroup = batch x 128-query tile (8 waves
// x 16 queries). Double-buffered LDS phi/g key-block tiles (one barrier per
// block, next block's global loads issued before compute). All WMMA loops are
// software-pipelined with explicit sched fences (cur/next fragment pairs).
// -----------------------------------------------------------------------------
__launch_bounds__(256)
__global__ void attn_kernel(const unsigned short* __restrict__ thetaT,
                            const unsigned short* __restrict__ phiT,
                            const unsigned short* __restrict__ gmat,
                            float* __restrict__ yT) {
  __shared__ __align__(16) unsigned short phi_lds[2][32][CI];    // [buf][key][c] 16 KB
  __shared__ __align__(16) unsigned short g_lds[2][CI][32];      // [buf][c][key] 16 KB
  __shared__ __align__(16) unsigned short P_lds[8][16][32];      // per-wave [q][key] 8 KB

  const int b    = blockIdx.y;
  const int tid  = threadIdx.x;
  const int wave = tid >> 5;
  const int lane = tid & 31;
  const int h    = lane >> 4;
  const int l15  = lane & 15;
  const int q0   = blockIdx.x * 128 + wave * 16;

  // theta A-fragments for this wave's 16 queries: K=128 -> 4 chunks, b128 loads.
  FragAB aTh[4];
  {
    const char* rowp = (const char*)(thetaT + (size_t)(b * NPOS + q0 + l15) * CI);
    #pragma unroll
    for (int kb = 0; kb < 4; ++kb) {
      aTh[kb].q[0] = *(const uint4*)(rowp + kb * 64 + h * 32);
      aTh[kb].q[1] = *(const uint4*)(rowp + kb * 64 + h * 32 + 16);
    }
  }

  // staging roles (32B per thread per matrix)
  const int s_key = tid >> 3, s_part = tid & 7;   // phi: [key][part*16 c]
  const int s_c   = tid >> 1, s_half = tid & 1;   // g:   [c][half*16 keys]
  const uint4* phi_src = (const uint4*)(phiT + (size_t)(b * NPOS + s_key) * CI + s_part * 16);
  const uint4* g_src   = (const uint4*)(gmat + (size_t)(b * CI + s_c) * NPOS + s_half * 16);
  // per 32-key block advance: phi += 32*CI bf16 = 8192B = 512 uint4; g += 32 bf16 = 4 uint4

  v8f yacc[8];
  #pragma unroll
  for (int cb = 0; cb < 8; ++cb)
    #pragma unroll
    for (int r = 0; r < 8; ++r) yacc[cb][r] = 0.f;

  float mrun[8], lrun[8];
  #pragma unroll
  for (int r = 0; r < 8; ++r) { mrun[r] = -1e30f; lrun[r] = 0.f; }

  // prologue: stage block 0 into buffer 0
  {
    uint4 rp0 = phi_src[0], rp1 = phi_src[1];
    uint4 rg0 = g_src[0],   rg1 = g_src[1];
    phi_src += 512; g_src += 4;
    uint4* pd = (uint4*)(&phi_lds[0][s_key][s_part * 16]);
    pd[0] = rp0; pd[1] = rp1;
    uint4* gd = (uint4*)(&g_lds[0][s_c][s_half * 16]);
    gd[0] = rg0; gd[1] = rg1;
  }
  __syncthreads();

  for (int blk = 0; blk < NPOS / 32; ++blk) {
    const int cur = blk & 1;
    const bool more = (blk + 1) < (NPOS / 32);

    // issue next block's global loads now; consumed (ds_store) after compute
    uint4 rp0, rp1, rg0, rg1;
    if (more) {
      rp0 = phi_src[0]; rp1 = phi_src[1];
      rg0 = g_src[0];   rg1 = g_src[1];
      phi_src += 512; g_src += 4;
    }

    // ---- scores: two 16-key tiles, K=128; interleaved independent chains ----
    v8f sc[2];
    #pragma unroll
    for (int r = 0; r < 8; ++r) { sc[0][r] = 0.f; sc[1][r] = 0.f; }
    {
      // frag(step): nt = step&1 (row +16 => +4096B), kb = step>>1 (+kb*64B)
      const char* base0 = (const char*)&phi_lds[cur][l15][0] + h * 32;
      FragAB pc, pn;
      pc.q[0] = *(const uint4*)(base0);
      pc.q[1] = *(const uint4*)(base0 + 16);
      #pragma unroll
      for (int step = 0; step < 8; ++step) {
        const int nt = step & 1, kb = step >> 1;
        if (step < 7) {
          const int s1 = step + 1;
          const char* a2 = base0 + ((s1 & 1) ? 4096 : 0) + (s1 >> 1) * 64;
          pn.q[0] = *(const uint4*)(a2);
          pn.q[1] = *(const uint4*)(a2 + 16);
        }
        sc[nt] = __builtin_amdgcn_wmma_f32_16x16x32_bf16(false, aTh[kb].v, false, pc.v,
                                                         (short)0, sc[nt], false, false);
        pc = pn;
        SCHED_FENCE();
      }
    }

    // ---- online softmax (rows live in a 16-lane half per D-slot r) ----
    float scale[8];
    #pragma unroll
    for (int r = 0; r < 8; ++r) {
      float v = fmaxf(sc[0][r], sc[1][r]);
      v = fmaxf(v, __shfl_xor(v, 1));
      v = fmaxf(v, __shfl_xor(v, 2));
      v = fmaxf(v, __shfl_xor(v, 4));
      v = fmaxf(v, __shfl_xor(v, 8));
      const float mnew = fmaxf(mrun[r], v);
      scale[r] = __expf(mrun[r] - mnew);
      mrun[r]  = mnew;
      sc[0][r] = __expf(sc[0][r] - mnew);        // p, in place
      sc[1][r] = __expf(sc[1][r] - mnew);
      float s = sc[0][r] + sc[1][r];
      s += __shfl_xor(s, 1);
      s += __shfl_xor(s, 2);
      s += __shfl_xor(s, 4);
      s += __shfl_xor(s, 8);
      lrun[r] = lrun[r] * scale[r] + s;
    }
    #pragma unroll
    for (int cb = 0; cb < 8; ++cb)
      #pragma unroll
      for (int r = 0; r < 8; ++r) yacc[cb][r] *= scale[r];

    // P (D-layout) -> per-wave LDS [q][key] bf16, then reload as A-fragment.
    #pragma unroll
    for (int r = 0; r < 8; ++r) {
      P_lds[wave][r + 8 * h][l15]      = f2bf(sc[0][r]);
      P_lds[wave][r + 8 * h][16 + l15] = f2bf(sc[1][r]);
    }
    FragAB aP;
    {
      const char* base = (const char*)&P_lds[wave][l15][0] + h * 32;
      aP.q[0] = *(const uint4*)(base);
      aP.q[1] = *(const uint4*)(base + 16);
    }

    // ---- yT[16q x 128c] += P[16x32] * gT[32 x 16c], 8 independent chains ----
    {
      const char* gbase = (const char*)&g_lds[cur][l15][0] + h * 32;  // +cb*1024B per tile
      FragAB gc, gn;
      gc.q[0] = *(const uint4*)(gbase);
      gc.q[1] = *(const uint4*)(gbase + 16);
      #pragma unroll
      for (int cb = 0; cb < 8; ++cb) {
        if (cb < 7) {
          const char* a2 = gbase + (cb + 1) * 1024;
          gn.q[0] = *(const uint4*)(a2);
          gn.q[1] = *(const uint4*)(a2 + 16);
        }
        yacc[cb] = __builtin_amdgcn_wmma_f32_16x16x32_bf16(false, aP.v, false, gc.v,
                                                           (short)0, yacc[cb], false, false);
        gc = gn;
        SCHED_FENCE();
      }
    }

    // stage next block into the other buffer (global data should have landed)
    if (more) {
      uint4* pd = (uint4*)(&phi_lds[cur ^ 1][s_key][s_part * 16]);
      pd[0] = rp0; pd[1] = rp1;
      uint4* gd = (uint4*)(&g_lds[cur ^ 1][s_c][s_half * 16]);
      gd[0] = rg0; gd[1] = rg1;
    }
    __syncthreads();
  }

  // finalize: divide by softmax denom, store yT f32 [b][q][c]
  #pragma unroll
  for (int r = 0; r < 8; ++r) {
    const float inv = 1.f / lrun[r];
    #pragma unroll
    for (int cb = 0; cb < 8; ++cb)
      yT[(size_t)(b * NPOS + q0 + r + 8 * h) * CI + cb * 16 + l15] = yacc[cb][r] * inv;
  }
}

// -----------------------------------------------------------------------------
// Kernel 3: out = w_out * y + b_out via bf16 WMMA (K=128 -> 4 chunks).
// One 16x16 output tile per wave; yT is [n][c] f32 -> pack pairs on the fly.
// -----------------------------------------------------------------------------
__launch_bounds__(256)
__global__ void out_kernel(const float* __restrict__ yT,
                           const float* __restrict__ wo,
                           const float* __restrict__ bo,
                           float* __restrict__ outp) {
  const int b    = blockIdx.z;
  const int co0  = blockIdx.y * 16;
  const int tid  = threadIdx.x;
  const int wave = tid >> 5;
  const int lane = tid & 31;
  const int h    = lane >> 4;
  const int l15  = lane & 15;
  const int n0   = (blockIdx.x * 8 + wave) * 16;

  FragAB aW[4];
  const float* wrow = wo + (size_t)(co0 + l15) * CI;
  #pragma unroll
  for (int kb = 0; kb < 4; ++kb) {
    const float* f0 = wrow + kb * 32 + h * 16;
    const float* f1 = f0 + 8;
    #pragma unroll
    for (int j = 0; j < 4; ++j) {
      aW[kb].u[j]     = pack_bf2(f0[2 * j], f0[2 * j + 1]);
      aW[kb].u[4 + j] = pack_bf2(f1[2 * j], f1[2 * j + 1]);
    }
  }

  v8f acc;
  #pragma unroll
  for (int r = 0; r < 8; ++r) acc[r] = 0.f;

  const float* yrow = yT + (size_t)(b * NPOS + n0 + l15) * CI;
  #pragma unroll
  for (int kb = 0; kb < 4; ++kb) {
    FragAB bY;   // B-frag [32 ci x 16 n]
    const float* f = yrow + kb * 32 + h * 16;
    #pragma unroll
    for (int j = 0; j < 8; ++j)
      bY.u[j] = pack_bf2(f[2 * j], f[2 * j + 1]);
    acc = __builtin_amdgcn_wmma_f32_16x16x32_bf16(false, aW[kb].v, false, bY.v,
                                                  (short)0, acc, false, false);
  }

  #pragma unroll
  for (int r = 0; r < 8; ++r) {
    const int co = co0 + 8 * h + r;
    outp[(size_t)(b * CO + co) * NPOS + n0 + l15] = acc[r] + bo[co];
  }
}

// -----------------------------------------------------------------------------
extern "C" void kernel_launch(void* const* d_in, const int* in_sizes, int n_in,
                              void* d_out, int out_size, void* d_ws, size_t ws_size,
                              hipStream_t stream) {
  (void)in_sizes; (void)n_in; (void)out_size; (void)ws_size;
  const float* x  = (const float*)d_in[0];
  const float* wt = (const float*)d_in[1];
  const float* bt = (const float*)d_in[2];
  const float* wp = (const float*)d_in[3];
  const float* bp = (const float*)d_in[4];
  const float* wg = (const float*)d_in[5];
  const float* bg = (const float*)d_in[6];
  const float* wo = (const float*)d_in[7];
  const float* bo = (const float*)d_in[8];
  float* outp = (float*)d_out;

  char* ws = (char*)d_ws;
  unsigned short* thetaT = (unsigned short*)(ws);                      // 4 MB
  unsigned short* phiT   = (unsigned short*)(ws + 4u  * 1024 * 1024);  // 4 MB
  unsigned short* gmat   = (unsigned short*)(ws + 8u  * 1024 * 1024);  // 4 MB
  float*          yT     = (float*)        (ws + 12u * 1024 * 1024);   // 8 MB

  proj_kernel<<<dim3(NPOS / 128, BATCH), 256, 0, stream>>>(x, wt, bt, wp, bp, wg, bg,
                                                           thetaT, phiT, gmat);
  attn_kernel<<<dim3(NPOS / 128, BATCH), 256, 0, stream>>>(thetaT, phiT, gmat, yT);
  out_kernel<<<dim3(NPOS / 128, CO / 16, BATCH), 256, 0, stream>>>(yT, wo, bo, outp);
}
